// GATNet_26474178413327
// MI455X (gfx1250) — matrix operand order repaired and checked by reading, not compiled, not run on hardware
//
#include <hip/hip_runtime.h>
#include <hip/hip_bf16.h>

// ---------------------------------------------------------------------------
// Problem constants (match reference)
// ---------------------------------------------------------------------------
#define NUM_NET 5
#define NN      10000            // nodes per net
#define NE      160000           // edges per net
#define NEE     (NE + NN)        // edges incl. self loops
#define BB      512              // graphs per batch
#define FXD     62               // node feature dim
#define KX      64               // FXD padded to 32-multiple
#define FXT     954              // cell feature dim
#define KC      960              // FXT padded to 32-multiple
#define DD      128              // output dim / head channels
#define H1      10               // heads layer 1
#define HD      (H1 * DD)        // 1280

typedef unsigned int   u32;
typedef unsigned short u16;
typedef __bf16 v16bf __attribute__((ext_vector_type(16)));
typedef float  v8f   __attribute__((ext_vector_type(8)));

static const u32 ENC_NINF = 0x007FFFFFu;   // order-preserving encode of -inf

// ---------------------------------------------------------------------------
// Helpers
// ---------------------------------------------------------------------------
__device__ __forceinline__ u16 f2bf(float f) {
  u32 u = __float_as_uint(f);
  u32 r = u + 0x7FFFu + ((u >> 16) & 1u);     // round-to-nearest-even
  return (u16)(r >> 16);
}
__device__ __forceinline__ float bf2f(u16 b) {
  return __uint_as_float(((u32)b) << 16);
}
// order-preserving monotonic float<->uint map (for atomicMax on floats)
__device__ __forceinline__ u32 fenc(float f) {
  u32 u = __float_as_uint(f);
  return (u & 0x80000000u) ? ~u : (u | 0x80000000u);
}
__device__ __forceinline__ float fdec(u32 u) {
  return __uint_as_float((u & 0x80000000u) ? (u & 0x7FFFFFFFu) : ~u);
}
__device__ __forceinline__ float lrelu02(float x) { return x > 0.f ? x : 0.2f * x; }
__device__ __forceinline__ float eluf(float x)    { return x > 0.f ? x : (__expf(x) - 1.f); }

// ---------------------------------------------------------------------------
// bf16 WMMA GEMM: C[M x Nout] = act(A[M x Kp] * W[Nout x Kp]^T + bias)
// A, W row-major bf16, same padded K stride (Kp % 32 == 0).
// Each wave owns a 16(M) x 64(N) C tile: 4 accumulators, A fragment reused
// across 4 B fragments -> 2.5 b128 loads per WMMA. 8 waves per block.
// Fragment layouts per CDNA5 ISA 7.12.2 (wave32).
// ---------------------------------------------------------------------------
union FragBF { v16bf v; u32 u[8]; };
union FragF  { v8f   v; float f[8]; };

__global__ __launch_bounds__(256) void wmma_gemm_bf16(
    const u16* __restrict__ A, const u16* __restrict__ W,
    const float* __restrict__ bias,
    float* __restrict__ outF, u16* __restrict__ outB,
    int M, int Nout, int Kp, int act) {
  int lane  = threadIdx.x & 31;
  int wave  = threadIdx.x >> 5;
  int tilesN = Nout >> 6;                    // 64-wide N tiles
  int tilesM = M >> 4;
  int t = blockIdx.x * 8 + wave;
  if (t >= tilesM * tilesN) return;
  int tm = (t / tilesN) << 4;
  int tn = (t % tilesN) << 6;
  int half = lane >> 4;
  int l16  = lane & 15;
  const u16* arow = A + (size_t)(tm + l16) * Kp;      // A: lane holds row M=l16
  const u16* brow[4];
#pragma unroll
  for (int nb = 0; nb < 4; ++nb)                       // B: lane holds col N=l16
    brow[nb] = W + (size_t)(tn + nb * 16 + l16) * Kp;

  FragF acc[4];
#pragma unroll
  for (int nb = 0; nb < 4; ++nb)
#pragma unroll
    for (int r = 0; r < 8; ++r) acc[nb].f[r] = 0.f;

  for (int kk = 0; kk < Kp; kk += 32) {
    FragBF a;
#pragma unroll
    for (int r = 0; r < 8; ++r) {
      // A 16x32 bf16: VGPR r holds K pair (r<4?0:16) + half*8 + (r%4)*2
      int ka = kk + ((r & 4) << 2) + half * 8 + ((r & 3) << 1);
      a.u[r] = *(const u32*)(arow + ka);
    }
#pragma unroll
    for (int nb = 0; nb < 4; ++nb) {
      FragBF b;
#pragma unroll
      for (int r = 0; r < 8; ++r) {
        // B 32x16 bf16: VGPR r holds K pair (r<4?0:16) + half*8 + (r%4)*2 (same as A)
        int kb = kk + ((r & 4) << 2) + half * 8 + ((r & 3) << 1);
        b.u[r] = *(const u32*)(brow[nb] + kb);
      }
      acc[nb].v = __builtin_amdgcn_wmma_f32_16x16x32_bf16(
          false, a.v, false, b.v, (short)0, acc[nb].v, false, false);
    }
  }

#pragma unroll
  for (int nb = 0; nb < 4; ++nb) {
    int n = tn + nb * 16 + l16;
    float bv = bias ? bias[n] : 0.f;
#pragma unroll
    for (int r = 0; r < 8; ++r) {
      int m = tm + half * 8 + r;      // C layout: VGPR r -> M = half*8 + r
      float v = acc[nb].f[r] + bv;
      if (act == 1)      v = v > 0.f ? v : 0.f;   // relu
      else if (act == 2) v = eluf(v);             // elu
      size_t idx = (size_t)m * Nout + n;
      if (outF) outF[idx] = v;
      if (outB) outB[idx] = f2bf(v);
    }
  }
}

// ---------------------------------------------------------------------------
// Small utility kernels
// ---------------------------------------------------------------------------
__global__ void fill_u32_k(u32* p, u32 v, int n) {
  int i = blockIdx.x * blockDim.x + threadIdx.x;
  if (i < n) p[i] = v;
}

// f32 [M x K] -> bf16 [M x Kp], zero padded
__global__ void cvt_pad_k(const float* __restrict__ src, u16* __restrict__ dst,
                          int M, int K, int Kp) {
  int i = blockIdx.x * blockDim.x + threadIdx.x;
  if (i >= M * Kp) return;
  int m = i / Kp, k = i - m * Kp;
  dst[i] = (k < K) ? f2bf(src[(size_t)m * K + k]) : (u16)0;
}

// ---------------------------------------------------------------------------
// CSR-by-destination build (edge list incl. implicit self loops)
// ---------------------------------------------------------------------------
__device__ __forceinline__ void edge_sd(const int* __restrict__ ei, int e,
                                        int& s, int& d) {
  if (e < NE) { s = ei[e]; d = ei[NE + e]; }
  else        { s = d = e - NE; }              // self loop
}

__global__ void count_k(const int* __restrict__ ei, int* __restrict__ counts) {
  int e = blockIdx.x * blockDim.x + threadIdx.x;
  if (e >= NEE) return;
  int s, d; edge_sd(ei, e, s, d);
  atomicAdd(&counts[d], 1);
}

// single-block exclusive scan of counts[NN] -> offsets[NN+1]
__global__ void scan_k(const int* __restrict__ counts, int* __restrict__ offsets) {
  __shared__ int part[256];
  int t = threadIdx.x;
  const int CH = (NN + 255) / 256;
  int base = t * CH;
  int s = 0;
  for (int j = 0; j < CH; ++j) {
    int idx = base + j;
    if (idx < NN) s += counts[idx];
  }
  part[t] = s;
  __syncthreads();
  for (int off = 1; off < 256; off <<= 1) {    // inclusive Hillis-Steele
    int v = (t >= off) ? part[t - off] : 0;
    __syncthreads();
    part[t] += v;
    __syncthreads();
  }
  int run = (t == 0) ? 0 : part[t - 1];
  for (int j = 0; j < CH; ++j) {
    int idx = base + j;
    if (idx < NN) { offsets[idx] = run; run += counts[idx]; }
  }
  if (t == 255) offsets[NN] = run;
}

__global__ void fill_csr_k(const int* __restrict__ ei,
                           const int* __restrict__ offsets,
                           int* __restrict__ cursor, int* __restrict__ esrc) {
  int e = blockIdx.x * blockDim.x + threadIdx.x;
  if (e >= NEE) return;
  int s, d; edge_sd(ei, e, s, d);
  int pos = offsets[d] + atomicAdd(&cursor[d], 1);
  esrc[pos] = s;
}

// ---------------------------------------------------------------------------
// GAT pieces (gather style, no float atomics in softmax/aggregation)
// ---------------------------------------------------------------------------
// al[n,h] = sum_c h[n,h,c]*a_s[h,c] ; ar likewise
__global__ void attn_logits_k(const u16* __restrict__ h,
                              const float* __restrict__ a_s,
                              const float* __restrict__ a_d,
                              float* __restrict__ al, float* __restrict__ ar,
                              int n, int H, int C) {
  int i = blockIdx.x * blockDim.x + threadIdx.x;
  if (i >= n * H) return;
  int node = i / H, hh = i - node * H;
  const u16* hp = h + (size_t)node * H * C + (size_t)hh * C;
  const float* as = a_s + hh * C;
  const float* ad = a_d + hh * C;
  float sa = 0.f, sd = 0.f;
  for (int c = 0; c < C; ++c) {
    float v = bf2f(hp[c]);
    sa += v * as[c];
    sd += v * ad[c];
  }
  al[i] = sa; ar[i] = sd;
}

// per (dst, head): segment softmax over incident edges -> coef[j*H+h]
__global__ void coef_k(const int* __restrict__ offsets,
                       const int* __restrict__ esrc,
                       const float* __restrict__ al,
                       const float* __restrict__ ar,
                       float* __restrict__ coef, int H) {
  int i = blockIdx.x * blockDim.x + threadIdx.x;
  if (i >= NN * H) return;
  int d = i / H, hh = i - d * H;
  int lo = offsets[d], hi = offsets[d + 1];
  float arv = ar[d * H + hh];
  float mx = -INFINITY;
  for (int j = lo; j < hi; ++j)
    mx = fmaxf(mx, lrelu02(al[esrc[j] * H + hh] + arv));
  float sum = 0.f;
  for (int j = lo; j < hi; ++j)
    sum += __expf(lrelu02(al[esrc[j] * H + hh] + arv) - mx);
  float inv = 1.f / sum;
  for (int j = lo; j < hi; ++j)
    coef[j * H + hh] = __expf(lrelu02(al[esrc[j] * H + hh] + arv) - mx) * inv;
}

// out[d,h,c8] = elu( sum_j coef * h[src,h,c8] + bias ) -> bf16 (layer 1)
__global__ void aggregate_bf16_k(const int* __restrict__ offsets,
                                 const int* __restrict__ esrc,
                                 const float* __restrict__ coef,
                                 const u16* __restrict__ h,
                                 const float* __restrict__ bias,
                                 u16* __restrict__ outB, int H, int C) {
  int chunks = C >> 3;
  int i = blockIdx.x * blockDim.x + threadIdx.x;
  if (i >= NN * H * chunks) return;
  int cc = i % chunks;
  int t  = i / chunks;
  int hh = t % H;
  int d  = t / H;
  int lo = offsets[d], hi = offsets[d + 1];
  float acc[8];
#pragma unroll
  for (int q = 0; q < 8; ++q) acc[q] = 0.f;
  for (int j = lo; j < hi; ++j) {
    float cf = coef[j * H + hh];
    const u16* hp = h + (size_t)esrc[j] * H * C + (size_t)hh * C + cc * 8;
#pragma unroll
    for (int q = 0; q < 8; ++q) acc[q] += cf * bf2f(hp[q]);
  }
  size_t ob = (size_t)d * H * C + (size_t)hh * C + cc * 8;
  const float* bp = bias + hh * C + cc * 8;
#pragma unroll
  for (int q = 0; q < 8; ++q) outB[ob + q] = f2bf(eluf(acc[q] + bp[q]));
}

// layer 2 (H==1) aggregation fused with global max pool into encoded g
__global__ void aggregate_pool_k(const int* __restrict__ offsets,
                                 const int* __restrict__ esrc,
                                 const float* __restrict__ coef,
                                 const u16* __restrict__ h,
                                 const float* __restrict__ bias,
                                 const int* __restrict__ batch,
                                 u32* __restrict__ g, int C) {
  int chunks = C >> 3;
  int i = blockIdx.x * blockDim.x + threadIdx.x;
  if (i >= NN * chunks) return;
  int cc = i % chunks;
  int d  = i / chunks;
  int lo = offsets[d], hi = offsets[d + 1];
  float acc[8];
#pragma unroll
  for (int q = 0; q < 8; ++q) acc[q] = 0.f;
  for (int j = lo; j < hi; ++j) {
    float cf = coef[j];
    const u16* hp = h + (size_t)esrc[j] * C + cc * 8;
#pragma unroll
    for (int q = 0; q < 8; ++q) acc[q] += cf * bf2f(hp[q]);
  }
  u32* gp = g + (size_t)batch[d] * C + cc * 8;
  const float* bp = bias + cc * 8;
#pragma unroll
  for (int q = 0; q < 8; ++q)
    atomicMax(&gp[q], fenc(eluf(acc[q] + bp[q])));
}

__global__ void dec_cvt_k(const u32* __restrict__ g, u16* __restrict__ gb, int n) {
  int i = blockIdx.x * blockDim.x + threadIdx.x;
  if (i < n) gb[i] = f2bf(fdec(g[i]));
}

__global__ void axpy02_k(const float* __restrict__ src, float* __restrict__ dst, int n) {
  int i = blockIdx.x * blockDim.x + threadIdx.x;
  if (i < n) dst[i] += 0.2f * src[i];
}

// ---------------------------------------------------------------------------
// L2 normalization / concat / final layer
// ---------------------------------------------------------------------------
__global__ void l2norm_rows_k(const float* __restrict__ src, u16* __restrict__ dst,
                              int K, int Kp) {
  __shared__ float red[256];
  int row = blockIdx.x;
  const float* rp = src + (size_t)row * K;
  float s = 0.f;
  for (int k = threadIdx.x; k < K; k += 256) { float v = rp[k]; s += v * v; }
  red[threadIdx.x] = s; __syncthreads();
  for (int off = 128; off > 0; off >>= 1) {
    if (threadIdx.x < off) red[threadIdx.x] += red[threadIdx.x + off];
    __syncthreads();
  }
  float inv = 1.f / fmaxf(sqrtf(red[0]), 1e-12f);
  u16* dp = dst + (size_t)row * Kp;
  for (int k = threadIdx.x; k < Kp; k += 256)
    dp[k] = (k < K) ? f2bf(rp[k] * inv) : (u16)0;
}

__global__ void concat_l2norm_k(const float* __restrict__ x11,
                                const float* __restrict__ x22,
                                const float* __restrict__ c3,
                                u16* __restrict__ dst) {
  __shared__ float row[512];
  __shared__ float red[256];
  int b = blockIdx.x, t = threadIdx.x;
  for (int k = t; k < 512; k += 256) {
    float v;
    if (k < 128)      v = x11[b * 128 + k];
    else if (k < 256) v = x22[b * 128 + (k - 128)];
    else              v = c3[b * 256 + (k - 256)];
    row[k] = v;
  }
  __syncthreads();
  float s = 0.f;
  for (int k = t; k < 512; k += 256) s += row[k] * row[k];
  red[t] = s; __syncthreads();
  for (int off = 128; off > 0; off >>= 1) {
    if (t < off) red[t] += red[t + off];
    __syncthreads();
  }
  float inv = 1.f / fmaxf(sqrtf(red[0]), 1e-12f);
  for (int k = t; k < 512; k += 256) dst[b * 512 + k] = f2bf(row[k] * inv);
}

__global__ void final_k(const float* __restrict__ f3, const float* __restrict__ Wo,
                        const float* __restrict__ bo, float* __restrict__ out) {
  int i = blockIdx.x * blockDim.x + threadIdx.x;
  if (i >= BB * 2) return;
  int b = i >> 1, o = i & 1;
  float s = bo[o];
  const float* fp = f3 + b * 128;
  const float* wp = Wo + o * 128;
  for (int c = 0; c < 128; ++c) s += fp[c] * wp[c];
  out[i] = s;
}

// ---------------------------------------------------------------------------
// Host orchestration
// ---------------------------------------------------------------------------
extern "C" void kernel_launch(void* const* d_in, const int* in_sizes, int n_in,
                              void* d_out, int out_size, void* d_ws, size_t ws_size,
                              hipStream_t stream) {
  (void)in_sizes; (void)n_in; (void)out_size; (void)ws_size;

  const float* x1_p    = (const float*)d_in[0];
  const int*   ei1_p   = (const int*)  d_in[1];
  const int*   batch1p = (const int*)  d_in[2];
  const float* x2_p    = (const float*)d_in[3];
  const int*   ei2_p   = (const int*)  d_in[4];
  const int*   batch2p = (const int*)  d_in[5];
  const float* cell_p  = (const float*)d_in[6];
  const float* W1_p  = (const float*)d_in[7];
  const float* a1s_p = (const float*)d_in[8];
  const float* a1d_p = (const float*)d_in[9];
  const float* b1_p  = (const float*)d_in[10];
  const float* W2_p  = (const float*)d_in[11];
  const float* a2s_p = (const float*)d_in[12];
  const float* a2d_p = (const float*)d_in[13];
  const float* b2_p  = (const float*)d_in[14];
  const float* Wg_p  = (const float*)d_in[15];
  const float* bg_p  = (const float*)d_in[16];
  const float* Wr1_p = (const float*)d_in[17];
  const float* br1_p = (const float*)d_in[18];
  const float* Wr2_p = (const float*)d_in[19];
  const float* br2_p = (const float*)d_in[20];
  const float* Wr3_p = (const float*)d_in[21];
  const float* br3_p = (const float*)d_in[22];
  const float* Wf1_p = (const float*)d_in[23];
  const float* bf1_p = (const float*)d_in[24];
  const float* Wf2_p = (const float*)d_in[25];
  const float* bf2_p = (const float*)d_in[26];
  const float* Wf3_p = (const float*)d_in[27];
  const float* bf3_p = (const float*)d_in[28];
  const float* Wo_p  = (const float*)d_in[29];
  const float* bo_p  = (const float*)d_in[30];

  // bump allocator over workspace
  char* ws = (char*)d_ws;
  size_t off = 0;
  auto alloc = [&](size_t bytes) -> void* {
    void* p = ws + off;
    off = (off + bytes + 255) & ~(size_t)255;
    return p;
  };

  u16*  xbf    = (u16*)alloc((size_t)NN * KX * 2);
  u16*  w1bf   = (u16*)alloc((size_t)HD * KX * 2);
  u16*  w2bf   = (u16*)alloc((size_t)DD * HD * 2);
  u16*  wgbf   = (u16*)alloc((size_t)DD * DD * 2);
  u16*  hbf    = (u16*)alloc((size_t)NN * HD * 2);   // layer1 linear transform
  u16*  h1e    = (u16*)alloc((size_t)NN * HD * 2);   // elu(aggregated layer1)
  u16*  h2bf   = (u16*)alloc((size_t)NN * DD * 2);   // layer2 linear transform
  float* al1   = (float*)alloc((size_t)NN * H1 * 4);
  float* ar1   = (float*)alloc((size_t)NN * H1 * 4);
  float* al2   = (float*)alloc((size_t)NN * 4);
  float* ar2   = (float*)alloc((size_t)NN * 4);
  int*  counts = (int*)alloc((size_t)NN * 4);
  int*  offs   = (int*)alloc((size_t)(NN + 1) * 4);
  int*  cursor = (int*)alloc((size_t)NN * 4);
  int*  esrc   = (int*)alloc((size_t)NEE * 4);
  float* coef1 = (float*)alloc((size_t)NEE * H1 * 4);
  float* coef2 = (float*)alloc((size_t)NEE * 4);
  u32*  g      = (u32*)alloc((size_t)BB * DD * 4);
  u16*  gbf    = (u16*)alloc((size_t)BB * DD * 2);
  float* gout  = (float*)alloc((size_t)BB * DD * 4);
  float* x11   = (float*)alloc((size_t)BB * DD * 4);
  float* x22   = (float*)alloc((size_t)BB * DD * 4);
  u16*  cnbf   = (u16*)alloc((size_t)BB * KC * 2);
  u16*  wr1bf  = (u16*)alloc((size_t)2048 * KC * 2);
  u16*  wr2bf  = (u16*)alloc((size_t)512 * 2048 * 2);
  u16*  wr3bf  = (u16*)alloc((size_t)256 * 512 * 2);
  u16*  c1bf   = (u16*)alloc((size_t)BB * 2048 * 2);
  u16*  c2bf   = (u16*)alloc((size_t)BB * 512 * 2);
  float* c3    = (float*)alloc((size_t)BB * 256 * 4);
  u16*  xcbf   = (u16*)alloc((size_t)BB * 512 * 2);
  u16*  wf1bf  = (u16*)alloc((size_t)2048 * 512 * 2);
  u16*  wf2bf  = (u16*)alloc((size_t)512 * 2048 * 2);
  u16*  wf3bf  = (u16*)alloc((size_t)128 * 512 * 2);
  u16*  f1bf   = (u16*)alloc((size_t)BB * 2048 * 2);
  u16*  f2bf   = (u16*)alloc((size_t)BB * 512 * 2);
  float* f3    = (float*)alloc((size_t)BB * 128 * 4);

  auto fill = [&](void* p, u32 v, int n) {
    fill_u32_k<<<(n + 255) / 256, 256, 0, stream>>>((u32*)p, v, n);
  };
  auto cvt = [&](const float* s, u16* d, int M, int K, int Kp) {
    int n = M * Kp;
    cvt_pad_k<<<(n + 255) / 256, 256, 0, stream>>>(s, d, M, K, Kp);
  };
  auto gemm = [&](const u16* A, const u16* W, const float* bias,
                  float* oF, u16* oB, int M, int Nout, int Kp, int act) {
    int tiles = (M / 16) * (Nout / 64);
    wmma_gemm_bf16<<<(tiles + 7) / 8, 256, 0, stream>>>(A, W, bias, oF, oB,
                                                        M, Nout, Kp, act);
  };

  hipMemsetAsync(x11, 0, (size_t)BB * DD * 4, stream);
  hipMemsetAsync(x22, 0, (size_t)BB * DD * 4, stream);

  for (int i = 0; i < NUM_NET; ++i) {
    cvt(W1_p + (size_t)i * HD * FXD, w1bf, HD, FXD, KX);
    cvt(W2_p + (size_t)i * DD * HD, w2bf, DD, HD, HD);
    cvt(Wg_p + (size_t)i * DD * DD, wgbf, DD, DD, DD);

    for (int side = 0; side < 2; ++side) {
      const float* x     = side ? x2_p + (size_t)i * NN * FXD : x1_p + (size_t)i * NN * FXD;
      const int*   ei    = side ? ei2_p + (size_t)i * 2 * NE  : ei1_p + (size_t)i * 2 * NE;
      const int*   batch = side ? batch2p + (size_t)i * NN    : batch1p + (size_t)i * NN;
      float*       xacc  = side ? x22 : x11;

      // ---- build CSR by destination (shared by both GAT layers) ----
      fill(counts, 0u, NN);
      fill(cursor, 0u, NN);
      count_k<<<(NEE + 255) / 256, 256, 0, stream>>>(ei, counts);
      scan_k<<<1, 256, 0, stream>>>(counts, offs);
      fill_csr_k<<<(NEE + 255) / 256, 256, 0, stream>>>(ei, offs, cursor, esrc);

      // ---- GAT layer 1 ----
      cvt(x, xbf, NN, FXD, KX);
      gemm(xbf, w1bf, nullptr, nullptr, hbf, NN, HD, KX, 0);       // h = x @ W1^T
      attn_logits_k<<<(NN * H1 + 255) / 256, 256, 0, stream>>>(
          hbf, a1s_p + (size_t)i * H1 * DD, a1d_p + (size_t)i * H1 * DD,
          al1, ar1, NN, H1, DD);
      coef_k<<<(NN * H1 + 255) / 256, 256, 0, stream>>>(offs, esrc, al1, ar1,
                                                        coef1, H1);
      {
        int n = NN * H1 * (DD / 8);
        aggregate_bf16_k<<<(n + 255) / 256, 256, 0, stream>>>(
            offs, esrc, coef1, hbf, b1_p + (size_t)i * HD, h1e, H1, DD);
      }

      // ---- GAT layer 2 ----
      gemm(h1e, w2bf, nullptr, nullptr, h2bf, NN, DD, HD, 0);
      attn_logits_k<<<(NN + 255) / 256, 256, 0, stream>>>(
          h2bf, a2s_p + (size_t)i * DD, a2d_p + (size_t)i * DD, al2, ar2, NN, 1, DD);
      coef_k<<<(NN + 255) / 256, 256, 0, stream>>>(offs, esrc, al2, ar2, coef2, 1);
      fill(g, ENC_NINF, BB * DD);
      {
        int n = NN * (DD / 8);
        aggregate_pool_k<<<(n + 255) / 256, 256, 0, stream>>>(
            offs, esrc, coef2, h2bf, b2_p + (size_t)i * DD, batch, g, DD);
      }

      // ---- gate ----
      dec_cvt_k<<<(BB * DD + 255) / 256, 256, 0, stream>>>(g, gbf, BB * DD);
      gemm(gbf, wgbf, bg_p + (size_t)i * DD, gout, nullptr, BB, DD, DD, 1);
      axpy02_k<<<(BB * DD + 255) / 256, 256, 0, stream>>>(gout, xacc, BB * DD);
    }
  }

  // ---- cell reduction MLP ----
  l2norm_rows_k<<<BB, 256, 0, stream>>>(cell_p, cnbf, FXT, KC);
  cvt(Wr1_p, wr1bf, 2048, FXT, KC);
  cvt(Wr2_p, wr2bf, 512, 2048, 2048);
  cvt(Wr3_p, wr3bf, 256, 512, 512);
  gemm(cnbf, wr1bf, br1_p, nullptr, c1bf, BB, 2048, KC, 1);
  gemm(c1bf, wr2bf, br2_p, nullptr, c2bf, BB, 512, 2048, 1);
  gemm(c2bf, wr3bf, br3_p, c3, nullptr, BB, 256, 512, 1);

  // ---- head MLP ----
  concat_l2norm_k<<<BB, 256, 0, stream>>>(x11, x22, c3, xcbf);
  cvt(Wf1_p, wf1bf, 2048, 512, 512);
  cvt(Wf2_p, wf2bf, 512, 2048, 2048);
  cvt(Wf3_p, wf3bf, 128, 512, 512);
  gemm(xcbf, wf1bf, bf1_p, nullptr, f1bf, BB, 2048, 512, 1);
  gemm(f1bf, wf2bf, bf2_p, nullptr, f2bf, BB, 512, 2048, 1);
  gemm(f2bf, wf3bf, bf3_p, f3, nullptr, BB, 128, 512, 1);
  final_k<<<(BB * 2 + 255) / 256, 256, 0, stream>>>(f3, Wo_p, bo_p, (float*)d_out);
}
